// PACKDETPostProcessor_71665824301808
// MI455X (gfx1250) — compile-verified, matrix operand order checked by reading, not run on hardware
//
#include <hip/hip_runtime.h>
#include <hip/hip_fp16.h>

// Problem constants (fixed by the reference harness)
#define N_IMG    8
#define C_CLS    80
#define H_DIM    168
#define W_DIM    256
#define L_LOC    (H_DIM * W_DIM)     // 43008
#define TILE_L   1024
#define N_TILES  (L_LOC / TILE_L)    // 42 (exact)
#define N_SCALES 5
#define TOPK     1000
#define KREAL    (N_SCALES * TOPK)   // 5000
#define KCAP     5120                // padded: multiple of 32 and 16
#define WORDS    (KCAP / 32)         // 160
#define ROWT     (KCAP / 16)         // 320
#define NSORT    8192
#define POST_N   100
#define NEGF     (-1.0e9f)
#define NEG2     (-2.0e9f)
#define CAND_TH  (-2.9444389791664403f)   // logit(0.05)
#define NMS_TH   0.6f

typedef __attribute__((ext_vector_type(16))) _Float16 v16h;
typedef __attribute__((ext_vector_type(8)))  float    v8f;

__device__ __forceinline__ unsigned ldsOff(const void* p) {
    // Flat addresses of LDS objects carry the LDS byte offset in addr[31:0]
    return (unsigned)(size_t)p;
}
__device__ __forceinline__ void asyncLoadB128(const void* g, unsigned lds) {
    asm volatile("global_load_async_to_lds_b128 %0, %1, off"
                 :: "v"(lds), "v"(g) : "memory");
}
__device__ __forceinline__ void waitAsync() {
    asm volatile("s_wait_asynccnt 0x0" ::: "memory");
}
__device__ __forceinline__ void waitDS() {
    asm volatile("s_wait_dscnt 0x0" ::: "memory");
}
__device__ __forceinline__ float sigm(float x) { return 1.0f / (1.0f + __expf(-x)); }

__device__ __forceinline__ unsigned ballot32(bool p) {
#if __has_builtin(__builtin_amdgcn_ballot_w32)
    return __builtin_amdgcn_ballot_w32(p);
#else
    return (unsigned)__ballot(p);
#endif
}

// ---------------------------------------------------------------- K0: init
__global__ void k0_init(unsigned* hist, unsigned* cnt, float* candS, int* candI) {
    int t = blockIdx.x * blockDim.x + threadIdx.x;
    if (t < N_IMG * N_SCALES * 256) hist[t] = 0u;
    if (t < N_IMG * N_SCALES * 2)   cnt[t]  = 0u;
    if (t < N_IMG * KREAL) { candS[t] = NEGF; candI[t] = 0; }
}

// ------------------------------------------------- K1: score-bin histogram
__global__ void k1_hist(const float* __restrict__ cls, const float* __restrict__ ctr,
                        const int* __restrict__ scales, unsigned* __restrict__ hist) {
    __shared__ float    s_ctr[TILE_L];
    __shared__ int      s_scl[TILE_L];
    __shared__ unsigned s_hist[N_SCALES * 256];
    const int t = threadIdx.x, n = blockIdx.y, l0 = blockIdx.x * TILE_L;
    for (int i = t; i < N_SCALES * 256; i += 256) s_hist[i] = 0u;
    // CDNA5 async copy: stage centerness + scale tile into LDS (reused 80x)
    asyncLoadB128(ctr + (size_t)n * L_LOC + l0 + t * 4, ldsOff(&s_ctr[t * 4]));
    asyncLoadB128(scales + l0 + t * 4, ldsOff(&s_scl[t * 4]));
    waitAsync();
    __syncthreads();
    for (int k = 0; k < 4; k++) s_ctr[t * 4 + k] = sigm(s_ctr[t * 4 + k]);
    __syncthreads();
    for (int c = 0; c < C_CLS; c++) {
        const float4 v = reinterpret_cast<const float4*>(
            cls + ((size_t)(n * C_CLS + c)) * L_LOC + l0)[t];
        const float vv[4] = {v.x, v.y, v.z, v.w};
#pragma unroll
        for (int k = 0; k < 4; k++) {
            float x = vv[k];
            if (x > CAND_TH) {
                float sc = sigm(x) * s_ctr[t * 4 + k];
                unsigned bin = __float_as_uint(sc) >> 22;  // monotonic, <=253 for sc in (0,1)
                if (bin > 255u) bin = 255u;
                atomicAdd(&s_hist[s_scl[t * 4 + k] * 256 + bin], 1u);
            }
        }
    }
    __syncthreads();
    for (int i = t; i < N_SCALES * 256; i += 256)
        atomicAdd(&hist[(size_t)n * N_SCALES * 256 + i], s_hist[i]);
}

// ---------------------------------------- K2: per-(image,scale) bin threshold
__global__ void k2_thresh(const unsigned* __restrict__ hist, int* __restrict__ meta) {
    int g = threadIdx.x;
    if (g >= N_IMG * N_SCALES) return;
    unsigned cum = 0; int T = -1, nab = 0, ntk = 0;
    for (int b = 255; b >= 1; --b) {
        unsigned h = hist[(size_t)g * 256 + b];
        if (T < 0 && cum + h >= (unsigned)TOPK) { T = b; nab = (int)cum; ntk = TOPK - (int)cum; }
        cum += h;
    }
    if (T < 0) { T = 0; nab = (int)cum; ntk = TOPK - (int)cum; }
    meta[g * 4 + 0] = T; meta[g * 4 + 1] = nab; meta[g * 4 + 2] = ntk; meta[g * 4 + 3] = 0;
}

// -------------------------------------------------- K3: compact top-k by bin
__global__ void k3_compact(const float* __restrict__ cls, const float* __restrict__ ctr,
                           const int* __restrict__ scales, const int* __restrict__ meta,
                           int* __restrict__ cnt, float* __restrict__ candS,
                           int* __restrict__ candI) {
    __shared__ float s_ctr[TILE_L];
    __shared__ int   s_scl[TILE_L];
    const int t = threadIdx.x, n = blockIdx.y, l0 = blockIdx.x * TILE_L;
    asyncLoadB128(ctr + (size_t)n * L_LOC + l0 + t * 4, ldsOff(&s_ctr[t * 4]));
    asyncLoadB128(scales + l0 + t * 4, ldsOff(&s_scl[t * 4]));
    waitAsync();
    __syncthreads();
    for (int k = 0; k < 4; k++) s_ctr[t * 4 + k] = sigm(s_ctr[t * 4 + k]);
    __syncthreads();
    for (int c = 0; c < C_CLS; c++) {
        const float4 v = reinterpret_cast<const float4*>(
            cls + ((size_t)(n * C_CLS + c)) * L_LOC + l0)[t];
        const float vv[4] = {v.x, v.y, v.z, v.w};
#pragma unroll
        for (int k = 0; k < 4; k++) {
            float x = vv[k];
            if (x <= CAND_TH) continue;
            float sc = sigm(x) * s_ctr[t * 4 + k];
            unsigned bin = __float_as_uint(sc) >> 22;
            if (bin > 255u) bin = 255u;
            int s = s_scl[t * 4 + k];
            int g = n * N_SCALES + s;
            int T = meta[g * 4], nab = meta[g * 4 + 1], ntk = meta[g * 4 + 2];
            int l = l0 + t * 4 + k;
            int idx = l * C_CLS + c;
            if ((int)bin > T) {
                int p = atomicAdd(&cnt[g * 2], 1);
                if (p < TOPK) { candS[(size_t)g * TOPK + p] = sc; candI[(size_t)g * TOPK + p] = idx; }
            } else if ((int)bin == T) {
                int e = atomicAdd(&cnt[g * 2 + 1], 1);
                if (e < ntk) {
                    int p = nab + e;
                    candS[(size_t)g * TOPK + p] = sc; candI[(size_t)g * TOPK + p] = idx;
                }
            }
        }
    }
}

// ----------------------------------------------------------- K4a: box decode
__global__ void k4a_decode(const float* __restrict__ candS, const int* __restrict__ candI,
                           const float* __restrict__ reg, const float* __restrict__ locs,
                           const int* __restrict__ isz,
                           float* ux1, float* uy1, float* ux2, float* uy2,
                           float* uar, int* ulab, float* us) {
    int tgt = blockIdx.x * blockDim.x + threadIdx.x;
    if (tgt >= N_IMG * KREAL) return;
    int n = tgt / KREAL, slot = tgt % KREAL;
    float sc = candS[(size_t)n * KREAL + slot];
    int idx = candI[(size_t)n * KREAL + slot];
    int loc = idx / C_CLS, lab = idx % C_CLS + 1;
    float lx = locs[loc * 2 + 0], ly = locs[loc * 2 + 1];
    float hh = (float)isz[n * 2 + 0], ww = (float)isz[n * 2 + 1];
    float r0 = reg[((size_t)(n * 4 + 0)) * L_LOC + loc];
    float r1 = reg[((size_t)(n * 4 + 1)) * L_LOC + loc];
    float r2 = reg[((size_t)(n * 4 + 2)) * L_LOC + loc];
    float r3 = reg[((size_t)(n * 4 + 3)) * L_LOC + loc];
    float x1 = fminf(fmaxf(lx - r0, 0.f), ww - 1.f);
    float y1 = fminf(fmaxf(ly - r1, 0.f), hh - 1.f);
    float x2 = fminf(fmaxf(lx + r2, 0.f), ww - 1.f);
    float y2 = fminf(fmaxf(ly + r3, 0.f), hh - 1.f);
    // branch-free validity
    int valid = (int)(sc > NEGF * 0.5f) & (int)(x2 - x1 + 1.f >= 0.f) &
                (int)(y2 - y1 + 1.f >= 0.f);
    size_t o = (size_t)n * KCAP + slot;
    ux1[o] = x1; uy1[o] = y1; ux2[o] = x2; uy2[o] = y2;
    uar[o] = (x2 - x1 + 1.f) * (y2 - y1 + 1.f);
    ulab[o] = lab;
    us[o] = valid ? sqrtf(fmaxf(sc, 1e-12f)) : NEGF;
}

// ------------------------------------------- K4b: bitonic sort (desc) in LDS
__global__ void __launch_bounds__(1024)
k4b_sort(const float* us, const int* ulab,
         const float* ux1, const float* uy1, const float* ux2, const float* uy2,
         const float* uar,
         float* ss, int* slab, float* sx1, float* sy1, float* sx2, float* sy2,
         float* sar) {
    __shared__ float key[NSORT];
    __shared__ int   pay[NSORT];
    const int n = blockIdx.x, t = threadIdx.x;
    for (int i = t; i < NSORT; i += 1024) {
        if (i < KREAL) { key[i] = us[(size_t)n * KCAP + i]; pay[i] = i; }
        else           { key[i] = NEG2; pay[i] = -1; }
    }
    __syncthreads();
    for (int k = 2; k <= NSORT; k <<= 1) {
        for (int j = k >> 1; j > 0; j >>= 1) {
            for (int i = t; i < NSORT; i += 1024) {
                int ixj = i ^ j;
                if (ixj > i) {
                    bool up = ((i & k) == 0);
                    float a = key[i], b = key[ixj];
                    bool sw = up ? (a < b) : (a > b);  // descending network
                    if (sw) {
                        key[i] = b; key[ixj] = a;
                        int pa = pay[i]; pay[i] = pay[ixj]; pay[ixj] = pa;
                    }
                }
            }
            __syncthreads();
        }
    }
    for (int i = t; i < KCAP; i += 1024) {
        float kv = key[i]; int p = pay[i];
        size_t o = (size_t)n * KCAP + i;
        if (p < 0 || kv < -1.5e9f) {
            ss[o] = NEG2; slab[o] = -1;
            sx1[o] = 0.f; sy1[o] = 0.f; sx2[o] = 0.f; sy2[o] = 0.f; sar[o] = 1.f;
        } else {
            size_t u = (size_t)n * KCAP + p;
            ss[o] = kv; slab[o] = ulab[u];
            sx1[o] = ux1[u]; sy1[o] = uy1[u]; sx2[o] = ux2[u]; sy2[o] = uy2[u];
            sar[o] = uar[u];
        }
    }
}

// ---------------- K5: suppression bitmask, 16x32 tile per wave, WMMA for the
// pairwise union term: D[m][n] = area_i[m]+area_j[n] via rank-2 f16 matmul.
// Predicates are fully branch-free (bitwise &, labels preloaded) so each test
// lowers to straight-line v_cmp/v_and feeding the wave32 ballot.
__global__ void __launch_bounds__(32)
k5_sup(const float* __restrict__ sx1, const float* __restrict__ sy1,
       const float* __restrict__ sx2, const float* __restrict__ sy2,
       const float* __restrict__ sar, const int* __restrict__ slab,
       unsigned* __restrict__ sup) {
    const int lane = threadIdx.x;
    const int n = blockIdx.z;
    const int rowBase = blockIdx.y * 16;
    const int wordIdx = blockIdx.x;
    const int colBase = wordIdx * 32;
    unsigned* rowOut = sup + ((size_t)n * KCAP + rowBase) * WORDS + wordIdx;
    if (colBase + 31 < rowBase) {           // strictly lower triangle: unused bits
        if (lane < 16) rowOut[(size_t)lane * WORDS] = 0u;
        return;
    }
    __shared__ float rX1[16], rY1[16], rX2[16], rY2[16], rAr[16];
    __shared__ int   rLb[16];
    __shared__ float cX1[32], cY1[32], cX2[32], cY2[32], cAr[32];
    __shared__ int   cLb[32];
    const size_t base = (size_t)n * KCAP;
    if (lane < 16) {
        size_t ri = base + rowBase + lane;
        rX1[lane] = sx1[ri]; rY1[lane] = sy1[ri]; rX2[lane] = sx2[ri]; rY2[lane] = sy2[ri];
        rAr[lane] = sar[ri]; rLb[lane] = slab[ri];
    }
    {
        size_t ci = base + colBase + lane;
        cX1[lane] = sx1[ci]; cY1[lane] = sy1[ci]; cX2[lane] = sx2[ci]; cY2[lane] = sy2[ci];
        cAr[lane] = sar[ci]; cLb[lane] = slab[ci];
    }
    __syncthreads();
    const int hi = lane >> 4, nn = lane & 15;
    const float SCL = 1.0f / 65536.0f;      // keep areas in f16 range
    // Per-lane column data in registers (no LDS access inside the r-loop for cols)
    const float c0x1 = cX1[nn],     c0y1 = cY1[nn],     c0x2 = cX2[nn],     c0y2 = cY2[nn];
    const float c1x1 = cX1[nn + 16], c1y1 = cY1[nn + 16], c1x2 = cX2[nn + 16], c1y2 = cY2[nn + 16];
    const int   cl0 = cLb[nn], cl1 = cLb[nn + 16];
    // A (16x32 f16): row m -> K0 = area_row[m]*SCL, K1 = 1 (K0,K1 in elements 0,1 of lanes 0-15; rest 0)
    v16h a = {};
    if (hi == 0) { a[0] = (_Float16)(rAr[nn] * SCL); a[1] = (_Float16)1.0f; }
    // B (32x16 f16): K0 row = 1, K1 row = area_col[n]*SCL
    v16h b0 = {}, b1 = {};
    if (hi == 0) {
        b0[0] = (_Float16)1.0f; b0[1] = (_Float16)(cAr[nn] * SCL);
        b1[0] = (_Float16)1.0f; b1[1] = (_Float16)(cAr[nn + 16] * SCL);
    }
    v8f z = {};
    v8f d0 = __builtin_amdgcn_wmma_f32_16x16x32_f16(false, a, false, b0, (short)0, z, false, false);
    v8f d1 = __builtin_amdgcn_wmma_f32_16x16x32_f16(false, a, false, b1, (short)0, z, false, false);
    unsigned bal0[8], bal1[8];
#pragma unroll
    for (int r = 0; r < 8; r++) {
        const int m = r + 8 * hi;   // D[m][n]: lanes 0-15 -> m=r, lanes 16-31 -> m=r+8
        const float rx1 = rX1[m], ry1 = rY1[m], rx2 = rX2[m], ry2 = rY2[m];
        const int rl = rLb[m];
        {   // cols colBase + [0,16)
            float iw = fmaxf(fminf(rx2, c0x2) - fmaxf(rx1, c0x1) + 1.0f, 0.f);
            float ih = fmaxf(fminf(ry2, c0y2) - fmaxf(ry1, c0y1) + 1.0f, 0.f);
            float inter = iw * ih;
            float uni = d0[r] * 65536.0f - inter;      // area_i+area_j from WMMA
            int pred = (int)(inter > NMS_TH * uni) & (int)(rl == cl0);
            bal0[r] = ballot32(pred != 0);
        }
        {   // cols colBase + [16,32)
            float iw = fmaxf(fminf(rx2, c1x2) - fmaxf(rx1, c1x1) + 1.0f, 0.f);
            float ih = fmaxf(fminf(ry2, c1y2) - fmaxf(ry1, c1y1) + 1.0f, 0.f);
            float inter = iw * ih;
            float uni = d1[r] * 65536.0f - inter;
            int pred = (int)(inter > NMS_TH * uni) & (int)(rl == cl1);
            bal1[r] = ballot32(pred != 0);
        }
    }
    // Assemble one u32 word per row. Ballot low 16 bits = rows r (lanes 0-15),
    // high 16 = rows r+8 (lanes 16-31); tile0 -> bits[15:0], tile1 -> bits[31:16].
    unsigned myword = 0;
#pragma unroll
    for (int r = 0; r < 8; r++) {
        unsigned wl = (bal0[r] & 0xFFFFu) | ((bal1[r] & 0xFFFFu) << 16);
        unsigned wh = (bal0[r] >> 16)     | (bal1[r] & 0xFFFF0000u);
        if (lane == r)     myword = wl;
        if (lane == r + 8) myword = wh;
    }
    if (lane < 16) rowOut[(size_t)lane * WORDS] = myword;
}

// --------------- K6: wave-synchronous greedy scan + emit top-100 (pre-sorted)
__global__ void __launch_bounds__(32)
k6_scan(const unsigned* __restrict__ sup, const float* __restrict__ ss,
        const int* __restrict__ slab,
        const float* __restrict__ sx1, const float* __restrict__ sy1,
        const float* __restrict__ sx2, const float* __restrict__ sy2,
        float* __restrict__ out) {
    __shared__ unsigned bits[WORDS];
    __shared__ int keepList[POST_N];
    const int lane = threadIdx.x, n = blockIdx.x;
    for (int w = lane; w < WORDS; w += 32) bits[w] = 0u;
    __syncthreads();
    int kcount = 0;
    for (int i = 0; i < KREAL && kcount < POST_N; i++) {
        size_t o = (size_t)n * KCAP + i;
        bool sup_i = (bits[i >> 5] >> (i & 31)) & 1u;
        bool keep = (ss[o] > NEGF * 0.5f) && !sup_i;    // uniform across wave
        if (keep) {
            if (lane == 0) keepList[kcount] = i;
            kcount++;
            const unsigned* row = sup + ((size_t)n * KCAP + i) * WORDS;
            for (int w = lane; w < WORDS; w += 32) bits[w] |= row[w];
            waitDS();
            __builtin_amdgcn_wave_barrier();
        }
    }
    __syncthreads();
    for (int o2 = lane; o2 < POST_N; o2 += 32) {
        float x1 = 0, y1 = 0, x2 = 0, y2 = 0, scv = 0, lbv = 0, vv = 0;
        if (o2 < kcount) {
            int i = keepList[o2];
            size_t o = (size_t)n * KCAP + i;
            x1 = sx1[o]; y1 = sy1[o]; x2 = sx2[o]; y2 = sy2[o];
            scv = ss[o]; lbv = (float)slab[o]; vv = 1.0f;
        }
        float* dets = out + (size_t)n * POST_N * 5 + o2 * 5;
        dets[0] = x1; dets[1] = y1; dets[2] = x2; dets[3] = y2; dets[4] = scv;
        out[(size_t)N_IMG * POST_N * 5 + (size_t)n * POST_N + o2] = lbv;   // labels
        out[(size_t)N_IMG * POST_N * 6 + (size_t)n * POST_N + o2] = vv;    // valid
    }
}

// ---------------------------------------------------------------- launcher
extern "C" void kernel_launch(void* const* d_in, const int* in_sizes, int n_in,
                              void* d_out, int out_size, void* d_ws, size_t ws_size,
                              hipStream_t stream) {
    const float* cls    = (const float*)d_in[0];
    const float* reg    = (const float*)d_in[1];
    const float* ctr    = (const float*)d_in[2];
    const float* locs   = (const float*)d_in[3];
    const int*   scales = (const int*)d_in[4];
    const int*   isz    = (const int*)d_in[5];
    float* out = (float*)d_out;
    char* ws = (char*)d_ws;
    size_t off = 0;
    auto alloc = [&](size_t b) { size_t o = off; off += (b + 255) & ~(size_t)255; return o; };

    unsigned* hist  = (unsigned*)(ws + alloc((size_t)N_IMG * N_SCALES * 256 * 4));
    int*      meta  = (int*)     (ws + alloc((size_t)N_IMG * N_SCALES * 4 * 4));
    int*      cnt   = (int*)     (ws + alloc((size_t)N_IMG * N_SCALES * 2 * 4));
    float*    candS = (float*)   (ws + alloc((size_t)N_IMG * KREAL * 4));
    int*      candI = (int*)     (ws + alloc((size_t)N_IMG * KREAL * 4));
    const size_t soa = (size_t)N_IMG * KCAP * 4;
    float* ux1 = (float*)(ws + alloc(soa));
    float* uy1 = (float*)(ws + alloc(soa));
    float* ux2 = (float*)(ws + alloc(soa));
    float* uy2 = (float*)(ws + alloc(soa));
    float* uar = (float*)(ws + alloc(soa));
    int*   ulab = (int*) (ws + alloc(soa));
    float* us  = (float*)(ws + alloc(soa));
    float* ss  = (float*)(ws + alloc(soa));
    int*   slab = (int*) (ws + alloc(soa));
    float* sx1 = (float*)(ws + alloc(soa));
    float* sy1 = (float*)(ws + alloc(soa));
    float* sx2 = (float*)(ws + alloc(soa));
    float* sy2 = (float*)(ws + alloc(soa));
    float* sar = (float*)(ws + alloc(soa));
    unsigned* sup = (unsigned*)(ws + alloc((size_t)N_IMG * KCAP * WORDS * 4));

    (void)in_sizes; (void)n_in; (void)out_size; (void)ws_size;

    k0_init<<<dim3((N_IMG * KREAL + 255) / 256), 256, 0, stream>>>(hist, (unsigned*)cnt, candS, candI);
    k1_hist<<<dim3(N_TILES, N_IMG), 256, 0, stream>>>(cls, ctr, scales, hist);
    k2_thresh<<<1, 64, 0, stream>>>(hist, meta);
    k3_compact<<<dim3(N_TILES, N_IMG), 256, 0, stream>>>(cls, ctr, scales, meta, cnt, candS, candI);
    k4a_decode<<<dim3((N_IMG * KREAL + 255) / 256), 256, 0, stream>>>(
        candS, candI, reg, locs, isz, ux1, uy1, ux2, uy2, uar, ulab, us);
    k4b_sort<<<dim3(N_IMG), 1024, 0, stream>>>(us, ulab, ux1, uy1, ux2, uy2, uar,
                                               ss, slab, sx1, sy1, sx2, sy2, sar);
    k5_sup<<<dim3(WORDS, ROWT, N_IMG), 32, 0, stream>>>(sx1, sy1, sx2, sy2, sar, slab, sup);
    k6_scan<<<dim3(N_IMG), 32, 0, stream>>>(sup, ss, slab, sx1, sy1, sx2, sy2, out);
}
